// DynamicConv2d_77747497992447
// MI455X (gfx1250) — matrix operand order repaired and checked by reading, not compiled, not run on hardware
//
#include <hip/hip_runtime.h>

// ---------------------------------------------------------------------------
// DynamicConv2d for MI455X (gfx1250, wave32, WMMA)
//  B=32, C=256, H=W=56, K=7, G=2, Cmid=C/4=64
// ---------------------------------------------------------------------------

typedef __attribute__((ext_vector_type(16))) _Float16 v16h;
typedef __attribute__((ext_vector_type(8)))  float    v8f;

#define BB   32
#define CC   256
#define HH   56
#define WW   56
#define KK   7
#define CMID 64
#define NPIX 49          // 7*7 output pixels of the small convs
#define NPAD 64          // pixel count padded to 4 N-tiles of 16
#define K1   (CC*9)      // 2304, reduction dim of proj1 GEMM
#define K2   (CMID*9)    // 576,  reduction dim of proj2 GEMM

// ---------------------------------------------------------------------------
// 1) 8x8 average pool: x (B,C,56,56) -> pooled (B,C,7,7), fp32
// ---------------------------------------------------------------------------
__global__ void pool_kernel(const float* __restrict__ x, float* __restrict__ pooled) {
    const int bc = blockIdx.x;                       // b*C + c
    const float* plane = x + (size_t)bc * (HH * WW);
    __shared__ float s[HH * WW];
    for (int i = threadIdx.x; i < HH * WW; i += blockDim.x) s[i] = plane[i];
    __syncthreads();
    const int t = threadIdx.x;
    if (t < NPIX) {
        const int wy = t / KK, wx = t % KK;
        float sum = 0.f;
        #pragma unroll
        for (int i = 0; i < 8; ++i)
            #pragma unroll
            for (int j = 0; j < 8; ++j)
                sum += s[(wy * 8 + i) * WW + (wx * 8 + j)];
        pooled[(size_t)bc * NPIX + t] = sum * (1.0f / 64.0f);
    }
}

// ---------------------------------------------------------------------------
// 2) fp32 -> fp16 repack (weights viewed as row-major MxK matrices)
// ---------------------------------------------------------------------------
__global__ void f32_to_f16_kernel(const float* __restrict__ src,
                                  _Float16* __restrict__ dst, int n) {
    int i = blockIdx.x * blockDim.x + threadIdx.x;
    if (i < n) dst[i] = (_Float16)src[i];
}

// ---------------------------------------------------------------------------
// 3) im2col for proj1: pooled (B,C,7,7) -> Bcol1 [B][NPAD cols][K1] f16,
//    column-major-K so B-fragments load as contiguous 32B per lane.
// ---------------------------------------------------------------------------
__global__ void im2col1_kernel(const float* __restrict__ pooled,
                               _Float16* __restrict__ bcol) {
    const int total = BB * NPAD * K1;
    int id = blockIdx.x * blockDim.x + threadIdx.x;
    if (id >= total) return;
    const int k = id % K1;
    const int n = (id / K1) % NPAD;
    const int b = id / (K1 * NPAD);
    float v = 0.f;
    if (n < NPIX) {
        const int c = k / 9, r = k % 9;
        const int dy = r / 3, dx = r % 3;
        const int oy = n / KK, ox = n % KK;
        const int iy = oy + dy - 1, ix = ox + dx - 1;
        if (iy >= 0 && iy < KK && ix >= 0 && ix < KK)
            v = pooled[((size_t)(b * CC + c)) * NPIX + iy * KK + ix];
    }
    bcol[id] = (_Float16)v;
}

// ---------------------------------------------------------------------------
// 4) im2col for proj2: hmid (B,64,7,7) f32 -> Bcol2 [B][NPAD][K2] f16
// ---------------------------------------------------------------------------
__global__ void im2col2_kernel(const float* __restrict__ hmid,
                               _Float16* __restrict__ bcol) {
    const int total = BB * NPAD * K2;
    int id = blockIdx.x * blockDim.x + threadIdx.x;
    if (id >= total) return;
    const int k = id % K2;
    const int n = (id / K2) % NPAD;
    const int b = id / (K2 * NPAD);
    float v = 0.f;
    if (n < NPIX) {
        const int c = k / 9, r = k % 9;
        const int dy = r / 3, dx = r % 3;
        const int oy = n / KK, ox = n % KK;
        const int iy = oy + dy - 1, ix = ox + dx - 1;
        if (iy >= 0 && iy < KK && ix >= 0 && ix < KK)
            v = hmid[((size_t)(b * CMID + c)) * NPIX + iy * KK + ix];
    }
    bcol[id] = (_Float16)v;
}

// ---------------------------------------------------------------------------
// WMMA fragment loaders, (wave32); contiguous K order chosen by this code, same for A and B
//   A 16x32 f16: lane L: M=L%16, half=L/16.
//     dword d: K = half*16 + 2d {,+1}
//   B 32x16 f16: lane L: N=L%16, half=L/16; dword d: K = half*16 + 2d {,+1}
//     -> 16 K-contiguous halfs = 32 contiguous bytes in column-major-K storage.
// ---------------------------------------------------------------------------
__device__ inline v16h load_a_frag(const _Float16* __restrict__ A, int KS,
                                   int M, int k0, int half_) {
    union { v16h h; unsigned u[8]; } fa;
    const unsigned* p = (const unsigned*)(A + (size_t)M * KS);
    #pragma unroll
    for (int d = 0; d < 4; ++d)
        fa.u[d] = p[(k0 + half_ * 16 + 2 * d) >> 1];
    #pragma unroll
    for (int d = 0; d < 4; ++d)
        fa.u[4 + d] = p[(k0 + half_ * 16 + 8 + 2 * d) >> 1];
    return fa.h;
}

__device__ inline v16h load_b_frag(const _Float16* __restrict__ Bcol, int KS,
                                   int col, int k0, int half_) {
    union { v16h h; uint4 q[2]; } fb;
    const uint4* p = (const uint4*)(Bcol + (size_t)col * KS + k0 + half_ * 16);
    fb.q[0] = p[0];
    fb.q[1] = p[1];
    return fb.h;
}

// ---------------------------------------------------------------------------
// 5) proj1 GEMM (64 x K1) x (K1 x 49) per batch, + BN + exact GELU.
//    grid = B, block = 256 (8 waves); 16 output tiles of 16x16 per batch.
// ---------------------------------------------------------------------------
__global__ void gemm1_kernel(const _Float16* __restrict__ W1h,   // 64 x K1
                             const _Float16* __restrict__ Bcol1, // [B][NPAD][K1]
                             const float* __restrict__ gamma,
                             const float* __restrict__ beta,
                             float* __restrict__ hmid)           // [B][64][49]
{
    const int b    = blockIdx.x;
    const int tid  = threadIdx.x;
    const int wave = tid >> 5;
    const int lane = tid & 31;
    const int half_ = lane >> 4;
    const int lrow  = lane & 15;
    const _Float16* Bb = Bcol1 + (size_t)b * NPAD * K1;
    const float bn_rs = rsqrtf(1.0f + 1e-5f);

    for (int t = wave; t < 16; t += 8) {
        const int m0 = (t >> 2) * 16;
        const int n0 = (t & 3) * 16;
        const int M   = m0 + lrow;
        const int col = n0 + lrow;
        v8f acc = {};
        for (int k0 = 0; k0 < K1; k0 += 32) {
            __builtin_prefetch(Bb + (size_t)col * K1 + k0 + 128, 0, 1);
            v16h a  = load_a_frag(W1h, K1, M, k0, half_);
            v16h bb = load_b_frag(Bb, K1, col, k0, half_);
            acc = __builtin_amdgcn_wmma_f32_16x16x32_f16(
                false, a, false, bb, (short)0, acc, false, false);
        }
        const int N = n0 + lrow;
        if (N < NPIX) {
            #pragma unroll
            for (int v = 0; v < 8; ++v) {
                const int Mv = m0 + v + 8 * half_;
                float xv = acc[v] * (gamma[Mv] * bn_rs) + beta[Mv];
                float g  = 0.5f * xv * (1.0f + erff(xv * 0.70710678118654752f));
                hmid[((size_t)(b * CMID + Mv)) * NPIX + N] = g;
            }
        }
    }
}

// ---------------------------------------------------------------------------
// 6) proj2 GEMM (512 x K2) x (K2 x 49) per batch, + bias -> logits fp32.
//    grid = B*16, block = 256; each wave owns one 16x16 tile (128/batch).
// ---------------------------------------------------------------------------
__global__ void gemm2_kernel(const _Float16* __restrict__ W2h,   // 512 x K2
                             const _Float16* __restrict__ Bcol2, // [B][NPAD][K2]
                             const float* __restrict__ bias,
                             float* __restrict__ logits)         // [B][512][49]
{
    const int blk  = blockIdx.x;
    const int b    = blk >> 4;
    const int grp  = blk & 15;
    const int tid  = threadIdx.x;
    const int wave = tid >> 5;
    const int lane = tid & 31;
    const int half_ = lane >> 4;
    const int lrow  = lane & 15;
    const int t  = grp * 8 + wave;       // 0..127
    const int m0 = (t >> 2) * 16;
    const int n0 = (t & 3) * 16;
    const _Float16* Bb = Bcol2 + (size_t)b * NPAD * K2;

    const int M   = m0 + lrow;
    const int col = n0 + lrow;
    v8f acc = {};
    for (int k0 = 0; k0 < K2; k0 += 32) {
        __builtin_prefetch(Bb + (size_t)col * K2 + k0 + 128, 0, 1);
        v16h a  = load_a_frag(W2h, K2, M, k0, half_);
        v16h bb = load_b_frag(Bb, K2, col, k0, half_);
        acc = __builtin_amdgcn_wmma_f32_16x16x32_f16(
            false, a, false, bb, (short)0, acc, false, false);
    }
    const int N = n0 + lrow;
    if (N < NPIX) {
        #pragma unroll
        for (int v = 0; v < 8; ++v) {
            const int Mv = m0 + v + 8 * half_;
            logits[((size_t)(b * (CC * 2) + Mv)) * NPIX + N] = acc[v] + bias[Mv];
        }
    }
}

// ---------------------------------------------------------------------------
// 7) softmax over G=2 + mix kernel banks -> dyn (B,C,49) fp32
// ---------------------------------------------------------------------------
__global__ void softmax_dyn_kernel(const float* __restrict__ logits,
                                   const float* __restrict__ w_dyn,
                                   float* __restrict__ dyn) {
    const int total = BB * CC * NPIX;
    int id = blockIdx.x * blockDim.x + threadIdx.x;
    if (id >= total) return;
    const int t = id % NPIX;
    const int c = (id / NPIX) % CC;
    const int b = id / (NPIX * CC);
    const float s0 = logits[((size_t)(b * (CC * 2) + c)) * NPIX + t];
    const float s1 = logits[((size_t)(b * (CC * 2) + CC + c)) * NPIX + t];
    const float m  = fmaxf(s0, s1);
    const float e0 = expf(s0 - m);
    const float e1 = expf(s1 - m);
    const float inv = 1.0f / (e0 + e1);
    const float p0 = e0 * inv, p1 = e1 * inv;
    dyn[id] = p0 * w_dyn[c * NPIX + t] + p1 * w_dyn[CC * NPIX + c * NPIX + t];
}

// ---------------------------------------------------------------------------
// 8) per-sample depthwise 7x7 conv, pad 3. One block per (b,c) plane.
//    Plane staged with halo in LDS (62x62 = 15.4 KB of the 320 KB WGP LDS),
//    49 weights hoisted to registers, fp32 FMAs.
// ---------------------------------------------------------------------------
#define PW 62   // 56 + 2*3
__global__ void depthwise_kernel(const float* __restrict__ x,
                                 const float* __restrict__ dyn,
                                 float* __restrict__ out) {
    const int bc = blockIdx.x;                  // b*C + c
    const float* plane  = x   + (size_t)bc * (HH * WW);
    float*       oplane = out + (size_t)bc * (HH * WW);
    const float* wp     = dyn + (size_t)bc * NPIX;

    __shared__ float s[PW * PW];
    for (int i = threadIdx.x; i < PW * PW; i += blockDim.x) s[i] = 0.f;
    __syncthreads();
    for (int i = threadIdx.x; i < HH * WW; i += blockDim.x) {
        const int iy = i / WW, ix = i % WW;
        s[(iy + 3) * PW + (ix + 3)] = plane[i];
    }
    float w[NPIX];
    #pragma unroll
    for (int i = 0; i < NPIX; ++i) w[i] = wp[i];
    __syncthreads();

    for (int i = threadIdx.x; i < HH * WW; i += blockDim.x) {
        const int oy = i / WW, ox = i % WW;
        const float* srow = &s[oy * PW + ox];
        float acc = 0.f;
        #pragma unroll
        for (int u = 0; u < 7; ++u)
            #pragma unroll
            for (int v = 0; v < 7; ++v)
                acc = fmaf(srow[u * PW + v], w[u * 7 + v], acc);
        oplane[i] = acc;
    }
}

// ---------------------------------------------------------------------------
// Launch
// ---------------------------------------------------------------------------
extern "C" void kernel_launch(void* const* d_in, const int* in_sizes, int n_in,
                              void* d_out, int out_size, void* d_ws, size_t ws_size,
                              hipStream_t stream) {
    const float* x       = (const float*)d_in[0];  // (32,256,56,56)
    const float* w_dyn   = (const float*)d_in[1];  // (2,256,7,7)
    const float* proj1_w = (const float*)d_in[2];  // (64,256,3,3) -> 64 x 2304
    const float* gamma   = (const float*)d_in[3];  // (64,)
    const float* beta    = (const float*)d_in[4];  // (64,)
    const float* proj2_w = (const float*)d_in[5];  // (512,64,3,3) -> 512 x 576
    const float* proj2_b = (const float*)d_in[6];  // (512,)
    float* out = (float*)d_out;

    char* ws = (char*)d_ws;
    size_t off = 0;
    float*     pooled = (float*)(ws + off);     off += (size_t)BB * CC * NPIX * 4;      // 1.6 MB
    _Float16*  bcol1  = (_Float16*)(ws + off);  off += (size_t)BB * NPAD * K1 * 2;      // 9.4 MB
    _Float16*  w1h    = (_Float16*)(ws + off);  off += (size_t)CMID * K1 * 2;           // 0.3 MB
    float*     hmid   = (float*)(ws + off);     off += (size_t)BB * CMID * NPIX * 4;    // 0.4 MB
    _Float16*  bcol2  = (_Float16*)(ws + off);  off += (size_t)BB * NPAD * K2 * 2;      // 2.4 MB
    _Float16*  w2h    = (_Float16*)(ws + off);  off += (size_t)(CC * 2) * K2 * 2;       // 0.6 MB
    float*     logits = (float*)(ws + off);     off += (size_t)BB * (CC * 2) * NPIX * 4;// 3.2 MB
    float*     dyn    = (float*)(ws + off);     off += (size_t)BB * CC * NPIX * 4;      // 1.6 MB
    (void)off; (void)ws_size; (void)in_sizes; (void)n_in; (void)out_size;

    // 1) pool
    pool_kernel<<<BB * CC, 128, 0, stream>>>(x, pooled);
    // 2) repack weights to f16
    {
        const int n1 = CMID * K1;                 // 147456
        const int n2 = (CC * 2) * K2;             // 294912
        f32_to_f16_kernel<<<(n1 + 255) / 256, 256, 0, stream>>>(proj1_w, w1h, n1);
        f32_to_f16_kernel<<<(n2 + 255) / 256, 256, 0, stream>>>(proj2_w, w2h, n2);
    }
    // 3) im2col for proj1
    im2col1_kernel<<<(BB * NPAD * K1) / 256, 256, 0, stream>>>(pooled, bcol1);
    // 4) proj1 GEMM + BN + GELU (WMMA)
    gemm1_kernel<<<BB, 256, 0, stream>>>(w1h, bcol1, gamma, beta, hmid);
    // 5) im2col for proj2
    im2col2_kernel<<<(BB * NPAD * K2) / 256, 256, 0, stream>>>(hmid, bcol2);
    // 6) proj2 GEMM + bias (WMMA)
    gemm2_kernel<<<BB * 16, 256, 0, stream>>>(w2h, bcol2, proj2_b, logits);
    // 7) softmax over G=2 + dynamic kernel mix
    softmax_dyn_kernel<<<(BB * CC * NPIX) / 256 + 1, 256, 0, stream>>>(logits, w_dyn, dyn);
    // 8) depthwise 7x7 with per-sample kernels
    depthwise_kernel<<<BB * CC, 256, 0, stream>>>(x, dyn, out);
}
